// memda_net_8564164788347
// MI455X (gfx1250) — matrix operand order repaired and checked by reading, not compiled, not run on hardware
//
#include <hip/hip_runtime.h>

typedef __attribute__((ext_vector_type(16))) _Float16 v16h;
typedef __attribute__((ext_vector_type(8)))  float    v8f;

#define WMMA(a,b,c) __builtin_amdgcn_wmma_f32_16x16x32_f16(false,(a),false,(b),(short)0,(c),false,false)

#define Bb   16
#define Nn   1024
#define Cc   256
#define Tt   7
#define BN   (Bb*Nn)          // 16384
#define BNT  (BN*Tt)          // 114688
#define Dd   64
#define NDR  144              // 9 calls * 16
#define NDRP 160              // padded K for dp2 GEMM
#define DEC  512
#define OUTC 12

// Flip to 0 if the assembler rejects the async mnemonic.
#define USE_ASYNC_G2L 1

// ---------------- async global->LDS helpers (CDNA5, ASYNCcnt) ----------------

__device__ __forceinline__ void async_ld16(void* lds, const void* g) {
#if USE_ASYNC_G2L
  unsigned ldsoff = (unsigned)(size_t)lds;   // LDS aperture: offset lives in addr[31:0]
  asm volatile("global_load_async_to_lds_b128 %0, %1, off"
               :: "v"(ldsoff), "v"(g) : "memory");
#else
  *(uint4*)lds = *(const uint4*)g;
#endif
}

__device__ __forceinline__ void async_wait0() {
#if USE_ASYNC_G2L
  asm volatile("s_wait_asynccnt 0" ::: "memory");
#endif
}

// ---------------- WMMA fragment helpers (contiguous k order: lane half h holds k = 16h + i) ----------------

// A fragment: 16x32 f16 tile, row-major source, lda in halfs.
__device__ __forceinline__ v16h frag_a(const _Float16* __restrict__ A, int lda) {
  const int lane = threadIdx.x & 31;
  const int row  = lane & 15;
  const int kb   = (lane >> 4) << 4;           // 0 or 16
  const _Float16* p = A + row * lda + kb;
  v16h a;
#pragma unroll
  for (int i = 0; i < 16; ++i) a[i] = p[i];
  return a;
}

// B fragment from K-major matrix Bm[k][n], ldb in halfs (row stride over k).
__device__ __forceinline__ v16h frag_b_k(const _Float16* __restrict__ Bm, int ldb) {
  const int lane = threadIdx.x & 31;
  const int col  = lane & 15;
  const int kb   = (lane >> 4) << 4;           // 0 or 16
  v16h b;
#pragma unroll
  for (int i = 0; i < 16; ++i) b[i] = Bm[(kb + i) * ldb + col];
  return b;
}

// B fragment from N-major (weight row-major [out][in]) matrix W[n][k].
__device__ __forceinline__ v16h frag_b_t(const _Float16* __restrict__ W, int ldw) {
  const int lane = threadIdx.x & 31;
  const int col  = lane & 15;
  const int kb   = (lane >> 4) << 4;
  const _Float16* p = W + col * ldw + kb;
  v16h b;
#pragma unroll
  for (int i = 0; i < 16; ++i) b[i] = p[i];
  return b;
}

// ---------------- small conversion / packing kernels ----------------

__global__ void k_cvt(const float* __restrict__ s, _Float16* __restrict__ d, int n) {
  int i = blockIdx.x * 256 + threadIdx.x;
  if (i < n) d[i] = (_Float16)s[i];
}

__global__ void k_pack_mwmb(const float* __restrict__ mW, const float* __restrict__ mb,
                            _Float16* __restrict__ d) {
  int i = blockIdx.x * 256 + threadIdx.x;
  if (i >= 16 * Cc) return;
  int r = i >> 8, c = i & 255;
  float v = (r < 14) ? mW[r * Cc + c] : ((r == 14) ? mb[c] : 0.f);
  d[i] = (_Float16)v;
}

__global__ void k_pad_dp2(const float* __restrict__ s, _Float16* __restrict__ d) {
  int i = blockIdx.x * 256 + threadIdx.x;
  if (i >= Cc * NDRP) return;
  int r = i / NDRP, k = i % NDRP;
  d[i] = (k < NDR) ? (_Float16)s[r * NDR + k] : (_Float16)0.f;
}

__global__ void k_pack_w2(const float* __restrict__ s, _Float16* __restrict__ d) {
  int i = blockIdx.x * 256 + threadIdx.x;
  if (i >= 16 * DEC) return;
  int r = i / DEC, k = i % DEC;
  d[i] = (r < OUTC) ? (_Float16)s[r * DEC + k] : (_Float16)0.f;
}

// hidden_x [B,C,N] -> hm[(b*N+n)*T + 0][c]
__global__ void k_hm_t0(const float* __restrict__ hx, float* __restrict__ hm) {
  int i = blockIdx.x * 256 + threadIdx.x;
  if (i >= Bb * Cc * Nn) return;
  int n = i & (Nn - 1);
  int c = (i >> 10) & (Cc - 1);
  int b = i >> 18;
  hm[((size_t)(b * Nn + n) * Tt) * Cc + c] = hx[i];
}

// ---------------- Embed_Trans: 256 -> 64 relu -> 256 ----------------

__global__ __launch_bounds__(256) void k_embed(
    const float* __restrict__ emb, const _Float16* __restrict__ w1f, const float* __restrict__ b1,
    const _Float16* __restrict__ w2f, const float* __restrict__ b2, float* __restrict__ hm)
{
  extern __shared__ char smem[];
  _Float16* Xs = (_Float16*)smem;       // 128*256
  _Float16* Y1 = Xs + 128 * 256;        // 128*64
  const int tid  = threadIdx.x;
  const int tok0 = blockIdx.x * 128;
  const int l    = blockIdx.y;
  for (int idx = tid; idx < 128 * 256; idx += 256) {
    int tt = idx & 127, c = idx >> 7;
    int tok = tok0 + tt; int b = tok >> 10; int n = tok & (Nn - 1);
    Xs[tt * 256 + c] = (_Float16)emb[(((size_t)l * Bb + b) * Cc + c) * Nn + n];
  }
  __syncthreads();
  const int wave = tid >> 5, lane = tid & 31;
  const int col = lane & 15, rb = (lane >> 4) * 8;
  const int m0 = wave * 16;
  const _Float16* w1l = w1f + (size_t)l * 64 * 256;
  {
    v8f acc[4] = {};
    for (int kk = 0; kk < 256; kk += 32) {
      v16h a = frag_a(Xs + m0 * 256 + kk, 256);
#pragma unroll
      for (int nt = 0; nt < 4; ++nt)
        acc[nt] = WMMA(a, frag_b_t(w1l + (nt * 16) * 256 + kk, 256), acc[nt]);
    }
#pragma unroll
    for (int nt = 0; nt < 4; ++nt) {
      float b0 = b1[l * 64 + nt * 16 + col];
#pragma unroll
      for (int r = 0; r < 8; ++r)
        Y1[(m0 + rb + r) * 64 + nt * 16 + col] = (_Float16)fmaxf(acc[nt][r] + b0, 0.f);
    }
  }
  __syncthreads();
  const _Float16* w2l = w2f + (size_t)l * 256 * 64;
  {
    v8f acc[16] = {};
    for (int kk = 0; kk < 64; kk += 32) {
      v16h a = frag_a(Y1 + m0 * 64 + kk, 64);
#pragma unroll
      for (int nt = 0; nt < 16; ++nt)
        acc[nt] = WMMA(a, frag_b_t(w2l + (nt * 16) * 64 + kk, 64), acc[nt]);
    }
#pragma unroll
    for (int nt = 0; nt < 16; ++nt) {
      float b0 = b2[l * 256 + nt * 16 + col];
#pragma unroll
      for (int r = 0; r < 8; ++r) {
        int tok = tok0 + m0 + rb + r;
        hm[((size_t)tok * Tt + (l + 1)) * Cc + nt * 16 + col] = acc[nt][r] + b0;
      }
    }
  }
}

// ---------------- memory attention + cts (fused per 128-token tile) ----------------

__global__ __launch_bounds__(256) void k_memattn(
    const float* __restrict__ hm,
    const _Float16* __restrict__ mp1f, const float* __restrict__ mp1b,
    const _Float16* __restrict__ pmf,
    const _Float16* __restrict__ mp2f, const float* __restrict__ mp2b,
    const _Float16* __restrict__ dp1f, const float* __restrict__ dp1b,
    float* __restrict__ res, _Float16* __restrict__ cts16)
{
  extern __shared__ char smem[];
  _Float16* Xs = (_Float16*)smem;            // 128*256
  _Float16* Qs = Xs + 128 * 256;             // 128*64
  _Float16* As = Qs + 128 * 64;              // 128*128
  _Float16* Rs = As + 128 * 128;             // 128*64
  const int tid = threadIdx.x;
  const int u0  = blockIdx.x * 128;
  for (int idx = tid; idx < 128 * 256; idx += 256) {
    int c = idx & 255, tt = idx >> 8;
    Xs[tt * 256 + c] = (_Float16)hm[(size_t)(u0 + tt) * 256 + c];
  }
  __syncthreads();
  const int wave = tid >> 5, lane = tid & 31;
  const int col = lane & 15, rb = (lane >> 4) * 8;
  const int m0 = wave * 16;
  {
    v8f accC[4] = {}, accQ[4] = {};
    for (int kk = 0; kk < 256; kk += 32) {
      v16h a = frag_a(Xs + m0 * 256 + kk, 256);
#pragma unroll
      for (int nt = 0; nt < 4; ++nt) {
        accC[nt] = WMMA(a, frag_b_t(dp1f + (nt * 16) * 256 + kk, 256), accC[nt]);
        accQ[nt] = WMMA(a, frag_b_t(mp1f + (nt * 16) * 256 + kk, 256), accQ[nt]);
      }
    }
#pragma unroll
    for (int nt = 0; nt < 4; ++nt) {
      float bc = dp1b[nt * 16 + col], bq = mp1b[nt * 16 + col];
#pragma unroll
      for (int r = 0; r < 8; ++r) {
        cts16[(size_t)(u0 + m0 + rb + r) * Dd + nt * 16 + col] = (_Float16)(accC[nt][r] + bc);
        Qs[(m0 + rb + r) * 64 + nt * 16 + col] = (_Float16)(accQ[nt][r] + bq);
      }
    }
  }
  __syncthreads();
  {
    v8f sc[8] = {};
    for (int kk = 0; kk < 64; kk += 32) {
      v16h a = frag_a(Qs + m0 * 64 + kk, 64);
#pragma unroll
      for (int nt = 0; nt < 8; ++nt)
        sc[nt] = WMMA(a, frag_b_t(pmf + (nt * 16) * 64 + kk, 64), sc[nt]);
    }
#pragma unroll
    for (int r = 0; r < 8; ++r) {
      float mx = -1e30f;
#pragma unroll
      for (int nt = 0; nt < 8; ++nt) mx = fmaxf(mx, sc[nt][r]);
#pragma unroll
      for (int s = 1; s < 16; s <<= 1) mx = fmaxf(mx, __shfl_xor(mx, s, 32));
      float sum = 0.f;
#pragma unroll
      for (int nt = 0; nt < 8; ++nt) { float e = __expf(sc[nt][r] - mx); sc[nt][r] = e; sum += e; }
#pragma unroll
      for (int s = 1; s < 16; s <<= 1) sum += __shfl_xor(sum, s, 32);
      float inv = 1.f / sum;
#pragma unroll
      for (int nt = 0; nt < 8; ++nt)
        As[(m0 + rb + r) * 128 + nt * 16 + col] = (_Float16)(sc[nt][r] * inv);
    }
  }
  __syncthreads();
  {
    v8f acc[4] = {};
    for (int kk = 0; kk < 128; kk += 32) {
      v16h a = frag_a(As + m0 * 128 + kk, 128);
#pragma unroll
      for (int nt = 0; nt < 4; ++nt)
        acc[nt] = WMMA(a, frag_b_k(pmf + kk * 64 + nt * 16, 64), acc[nt]);
    }
#pragma unroll
    for (int nt = 0; nt < 4; ++nt)
#pragma unroll
      for (int r = 0; r < 8; ++r)
        Rs[(m0 + rb + r) * 64 + nt * 16 + col] = (_Float16)acc[nt][r];
  }
  __syncthreads();
  {
    v8f acc[16] = {};
    for (int kk = 0; kk < 64; kk += 32) {
      v16h a = frag_a(Rs + m0 * 64 + kk, 64);
#pragma unroll
      for (int nt = 0; nt < 16; ++nt)
        acc[nt] = WMMA(a, frag_b_t(mp2f + (nt * 16) * 64 + kk, 64), acc[nt]);
    }
#pragma unroll
    for (int nt = 0; nt < 16; ++nt) {
      float b0 = mp2b[nt * 16 + col];
#pragma unroll
      for (int r = 0; r < 8; ++r)
        res[(size_t)(u0 + m0 + rb + r) * 256 + nt * 16 + col] = acc[nt][r] + b0;
    }
  }
}

// ---------------- NTN: relu(cos(x1 W1[k], x2 W2[k]) + [x1|x2].V[k] + b[k]) ----------------

template<int DD>
__global__ __launch_bounds__(256) void k_ntn(
    const _Float16* __restrict__ cts16, int t1o, int t2o,
    const _Float16* __restrict__ W1, const _Float16* __restrict__ W2,
    const _Float16* __restrict__ Vf, const float* __restrict__ bb,
    _Float16* __restrict__ drifts16, int doff)
{
  extern __shared__ char smem[];
  _Float16* X1 = (_Float16*)smem;                  // 128*DD
  _Float16* X2 = X1 + 128 * DD;                    // 128*DD
  float*    SD = (float*)(X2 + 128 * DD);          // 128*16
  const int tid  = threadIdx.x;
  const int tok0 = blockIdx.x * 128;
  constexpr int CH = DD / 8;                       // 16-byte chunks per row
  for (int idx = tid; idx < 128 * CH; idx += 256) {
    int tt = idx / CH, ch = idx % CH;
    size_t gbase = (size_t)(tok0 + tt) * Tt * Dd;  // halfs
    async_ld16(X1 + tt * DD + ch * 8, cts16 + gbase + t1o * Dd + ch * 8);
    async_ld16(X2 + tt * DD + ch * 8, cts16 + gbase + t2o * Dd + ch * 8);
  }
  async_wait0();
  __syncthreads();
  const int wave = tid >> 5, lane = tid & 31;
  const int col = lane & 15, rb = (lane >> 4) * 8;
  const int m0 = wave * 16;
  constexpr int NT = DD / 16;

  // part2 for all 16 k at once: [x1|x2] (128 x 2DD) @ V^T (2DD x 16); k lands on lane-col
  v8f p2a = {};
  for (int kk = 0; kk < 2 * DD; kk += 32) {
    const _Float16* src = (kk < DD) ? (X1 + m0 * DD + kk) : (X2 + m0 * DD + kk - DD);
    v16h a = frag_a(src, DD);
    p2a = WMMA(a, frag_b_t(Vf + kk, 2 * DD), p2a);
  }

  for (int kidx = 0; kidx < 16; ++kidx) {
    const _Float16* W1k = W1 + (size_t)kidx * DD * DD;
    const _Float16* W2k = W2 + (size_t)kidx * DD * DD;
    v8f t1[NT] = {}, t2[NT] = {};
    for (int kk = 0; kk < DD; kk += 32) {
      v16h a1 = frag_a(X1 + m0 * DD + kk, DD);
      v16h a2 = frag_a(X2 + m0 * DD + kk, DD);
#pragma unroll
      for (int nt = 0; nt < NT; ++nt) {
        t1[nt] = WMMA(a1, frag_b_k(W1k + kk * DD + nt * 16, DD), t1[nt]);
        t2[nt] = WMMA(a2, frag_b_k(W2k + kk * DD + nt * 16, DD), t2[nt]);
      }
    }
    const float bk = bb[kidx];
#pragma unroll
    for (int r = 0; r < 8; ++r) {
      float dot = 0.f, n1 = 0.f, n2 = 0.f;
#pragma unroll
      for (int nt = 0; nt < NT; ++nt) {
        float av = t1[nt][r], bv = t2[nt][r];
        dot += av * bv; n1 += av * av; n2 += bv * bv;
      }
#pragma unroll
      for (int s = 1; s < 16; s <<= 1) {
        dot += __shfl_xor(dot, s, 32);
        n1  += __shfl_xor(n1,  s, 32);
        n2  += __shfl_xor(n2,  s, 32);
      }
      if (col == kidx) {
        float d1 = fmaxf(sqrtf(n1), 1e-8f), d2 = fmaxf(sqrtf(n2), 1e-8f);
        SD[(m0 + rb + r) * 16 + kidx] = fmaxf(dot / (d1 * d2) + p2a[r] + bk, 0.f);
      }
    }
  }
  __syncthreads();
  for (int idx = tid; idx < 128 * 16; idx += 256) {
    int tt = idx >> 4, k = idx & 15;
    drifts16[(size_t)(tok0 + tt) * NDR + doff + k] = (_Float16)SD[idx];
  }
}

// ---------------- drift projection + meta (Wm softmax, bm) ----------------

__global__ __launch_bounds__(256) void k_drift(
    const _Float16* __restrict__ drifts16,
    const _Float16* __restrict__ dp2f, const float* __restrict__ dp2b,
    const _Float16* __restrict__ mwf, const float* __restrict__ mWb, const float* __restrict__ mbb,
    float* __restrict__ wm15)
{
  extern __shared__ char smem[];
  _Float16* Ds = (_Float16*)smem;          // 128*160 (144 data + 16 zero pad)
  _Float16* Ms = Ds + 128 * NDRP;          // 128*256
  const int tid  = threadIdx.x;
  const int tok0 = blockIdx.x * 128;
  // 144 halfs = 288B = 18 chunks of 16B per row
  for (int idx = tid; idx < 128 * 18; idx += 256) {
    int tt = idx / 18, ch = idx % 18;
    async_ld16(Ds + tt * NDRP + ch * 8, drifts16 + (size_t)(tok0 + tt) * NDR + ch * 8);
  }
  for (int idx = tid; idx < 128 * 2; idx += 256) {   // zero the 16-half pad region
    int tt = idx >> 1, ch = idx & 1;
    uint4 z = {0, 0, 0, 0};
    *(uint4*)(Ds + tt * NDRP + NDR + ch * 8) = z;
  }
  async_wait0();
  __syncthreads();
  const int wave = tid >> 5, lane = tid & 31;
  const int col = lane & 15, rb = (lane >> 4) * 8;
  const int m0 = wave * 16;
  {
    v8f acc[16] = {};
    for (int kk = 0; kk < NDRP; kk += 32) {
      v16h a = frag_a(Ds + m0 * NDRP + kk, NDRP);
#pragma unroll
      for (int nt = 0; nt < 16; ++nt)
        acc[nt] = WMMA(a, frag_b_t(dp2f + (nt * 16) * NDRP + kk, NDRP), acc[nt]);
    }
#pragma unroll
    for (int nt = 0; nt < 16; ++nt) {
      float b0 = dp2b[nt * 16 + col];
#pragma unroll
      for (int r = 0; r < 8; ++r)
        Ms[(m0 + rb + r) * 256 + nt * 16 + col] = (_Float16)(acc[nt][r] + b0);
    }
  }
  __syncthreads();
  v8f acc = {};
  for (int kk = 0; kk < 256; kk += 32) {
    v16h a = frag_a(Ms + m0 * 256 + kk, 256);
    acc = WMMA(a, frag_b_t(mwf + kk, 256), acc);
  }
  float bias0 = (col < 14) ? mWb[col] : ((col == 14) ? mbb[0] : 0.f);
#pragma unroll
  for (int r = 0; r < 8; ++r) {
    float v  = acc[r] + bias0;
    float sv = (col < 14) ? v : -1e30f;
    float mx = sv;
#pragma unroll
    for (int s = 1; s < 16; s <<= 1) mx = fmaxf(mx, __shfl_xor(mx, s, 32));
    float e = (col < 14) ? __expf(v - mx) : 0.f;
    float sum = e;
#pragma unroll
    for (int s = 1; s < 16; s <<= 1) sum += __shfl_xor(sum, s, 32);
    float out = (col < 14) ? (e / sum) : v;
    if (col < 15)
      wm15[(size_t)(tok0 + m0 + rb + r) * 15 + col] = out;
  }
}

// ---------------- weighted merge + decoder ----------------

__global__ __launch_bounds__(256) void k_dec(
    const float* __restrict__ hm, const float* __restrict__ res, const float* __restrict__ wm15,
    const _Float16* __restrict__ w1f, const float* __restrict__ b1,
    const _Float16* __restrict__ w2f, const float* __restrict__ b2,
    float* __restrict__ out)
{
  extern __shared__ char smem[];
  _Float16* Hs = (_Float16*)smem;                     // 128*256
  _Float16* Ys = Hs + 128 * 256;                      // 128*512
  float*    Ws = (float*)(Ys + 128 * 512);            // 128*15
  const int tid  = threadIdx.x;
  const int tok0 = blockIdx.x * 128;
  for (int idx = tid; idx < 128 * 15; idx += 256)
    Ws[idx] = wm15[(size_t)tok0 * 15 + idx];
  __syncthreads();
  for (int idx = tid; idx < 128 * 256; idx += 256) {
    int c = idx & 255, tt = idx >> 8;
    const float* w = Ws + tt * 15;
    size_t base = ((size_t)(tok0 + tt) * Tt) * Cc + c;
    float acc = w[14];                                // bm
#pragma unroll
    for (int t = 0; t < Tt; ++t)
      acc += w[t] * hm[base + (size_t)t * Cc] + w[Tt + t] * res[base + (size_t)t * Cc];
    Hs[idx] = (_Float16)acc;
  }
  __syncthreads();
  const int wave = tid >> 5, lane = tid & 31;
  const int col = lane & 15, rb = (lane >> 4) * 8;
  const int m0 = wave * 16;
  for (int ng = 0; ng < 2; ++ng) {
    v8f acc[16] = {};
    for (int kk = 0; kk < 256; kk += 32) {
      v16h a = frag_a(Hs + m0 * 256 + kk, 256);
#pragma unroll
      for (int nt = 0; nt < 16; ++nt)
        acc[nt] = WMMA(a, frag_b_t(w1f + ((ng * 16 + nt) * 16) * 256 + kk, 256), acc[nt]);
    }
#pragma unroll
    for (int nt = 0; nt < 16; ++nt) {
      float b0 = b1[(ng * 16 + nt) * 16 + col];
#pragma unroll
      for (int r = 0; r < 8; ++r)
        Ys[(m0 + rb + r) * 512 + (ng * 16 + nt) * 16 + col] = (_Float16)fmaxf(acc[nt][r] + b0, 0.f);
    }
  }
  __syncthreads();
  v8f acc = {};
  for (int kk = 0; kk < 512; kk += 32) {
    v16h a = frag_a(Ys + m0 * 512 + kk, 512);
    acc = WMMA(a, frag_b_t(w2f + kk, 512), acc);
  }
  if (col < OUTC) {
    float b0 = b2[col];
#pragma unroll
    for (int r = 0; r < 8; ++r) {
      int tok = tok0 + m0 + rb + r;
      int b = tok >> 10, n = tok & (Nn - 1);
      out[((size_t)b * OUTC + col) * Nn + n] = acc[r] + b0;
    }
  }
}

// ---------------- host ----------------

extern "C" void kernel_launch(void* const* d_in, const int* in_sizes, int n_in,
                              void* d_out, int out_size, void* d_ws, size_t ws_size,
                              hipStream_t stream) {
  (void)in_sizes; (void)n_in; (void)out_size; (void)ws_size;
  const float* hidden_x  = (const float*)d_in[0];
  const float* embedding = (const float*)d_in[1];
  const float* et_w1 = (const float*)d_in[2];
  const float* et_b1 = (const float*)d_in[3];
  const float* et_w2 = (const float*)d_in[4];
  const float* et_b2 = (const float*)d_in[5];
  const float* pmem  = (const float*)d_in[6];
  const float* mp1_w = (const float*)d_in[7];
  const float* mp1_b = (const float*)d_in[8];
  const float* mp2_w = (const float*)d_in[9];
  const float* mp2_b = (const float*)d_in[10];
  const float* dp1_w = (const float*)d_in[11];
  const float* dp1_b = (const float*)d_in[12];
  const float* x1_V  = (const float*)d_in[13];
  const float* x1_W1 = (const float*)d_in[14];
  const float* x1_W2 = (const float*)d_in[15];
  const float* x1_b  = (const float*)d_in[16];
  const float* x2_V  = (const float*)d_in[17];
  const float* x2_W1 = (const float*)d_in[18];
  const float* x2_W2 = (const float*)d_in[19];
  const float* x2_b  = (const float*)d_in[20];
  const float* y_V   = (const float*)d_in[21];
  const float* y_W1  = (const float*)d_in[22];
  const float* y_W2  = (const float*)d_in[23];
  const float* y_b   = (const float*)d_in[24];
  const float* xy_V  = (const float*)d_in[25];
  const float* xy_W1 = (const float*)d_in[26];
  const float* xy_W2 = (const float*)d_in[27];
  const float* xy_b  = (const float*)d_in[28];
  const float* dp2_w = (const float*)d_in[29];
  const float* dp2_b = (const float*)d_in[30];
  const float* mW_w  = (const float*)d_in[31];
  const float* mW_b  = (const float*)d_in[32];
  const float* mb_w  = (const float*)d_in[33];
  const float* mb_b  = (const float*)d_in[34];
  const float* dec_w1 = (const float*)d_in[35];
  const float* dec_b1 = (const float*)d_in[36];
  const float* dec_w2 = (const float*)d_in[37];
  const float* dec_b2 = (const float*)d_in[38];

  char* p = (char*)d_ws;
  auto alloc = [&](size_t bytes) -> void* {
    void* r = (void*)p;
    p += (bytes + 255) & ~(size_t)255;
    return r;
  };
  float*    hm       = (float*)alloc((size_t)BNT * Cc * 4);
  float*    res      = (float*)alloc((size_t)BNT * Cc * 4);
  _Float16* cts16    = (_Float16*)alloc((size_t)BNT * Dd * 2);
  _Float16* drifts16 = (_Float16*)alloc((size_t)BN * NDR * 2);
  float*    wm15     = (float*)alloc((size_t)BN * 15 * 4);

  _Float16* et_w1f = (_Float16*)alloc(6 * 64 * 256 * 2);
  _Float16* et_w2f = (_Float16*)alloc(6 * 256 * 64 * 2);
  _Float16* mp1f   = (_Float16*)alloc(64 * 256 * 2);
  _Float16* pmf    = (_Float16*)alloc(128 * 64 * 2);
  _Float16* mp2f   = (_Float16*)alloc(256 * 64 * 2);
  _Float16* dp1f   = (_Float16*)alloc(64 * 256 * 2);
  _Float16* x1w1f  = (_Float16*)alloc(3 * 16 * 64 * 64 * 2);
  _Float16* x1w2f  = (_Float16*)alloc(3 * 16 * 64 * 64 * 2);
  _Float16* x2w1f  = (_Float16*)alloc(2 * 16 * 64 * 64 * 2);
  _Float16* x2w2f  = (_Float16*)alloc(2 * 16 * 64 * 64 * 2);
  _Float16* yw1f   = (_Float16*)alloc(2 * 16 * 64 * 64 * 2);
  _Float16* yw2f   = (_Float16*)alloc(2 * 16 * 64 * 64 * 2);
  _Float16* xyw1f  = (_Float16*)alloc(2 * 16 * 128 * 128 * 2);
  _Float16* xyw2f  = (_Float16*)alloc(2 * 16 * 128 * 128 * 2);
  _Float16* x1vf   = (_Float16*)alloc(3 * 16 * 128 * 2);
  _Float16* x2vf   = (_Float16*)alloc(2 * 16 * 128 * 2);
  _Float16* yvf    = (_Float16*)alloc(2 * 16 * 128 * 2);
  _Float16* xyvf   = (_Float16*)alloc(2 * 16 * 256 * 2);
  _Float16* dp2f   = (_Float16*)alloc(Cc * NDRP * 2);
  _Float16* mwmbf  = (_Float16*)alloc(16 * 256 * 2);
  _Float16* decw1f = (_Float16*)alloc(DEC * Cc * 2);
  _Float16* decw2f = (_Float16*)alloc(16 * DEC * 2);

  auto cvt = [&](const float* s, _Float16* d, int n) {
    k_cvt<<<(n + 255) / 256, 256, 0, stream>>>(s, d, n);
  };
  cvt(et_w1, et_w1f, 6 * 64 * 256);
  cvt(et_w2, et_w2f, 6 * 256 * 64);
  cvt(mp1_w, mp1f, 64 * 256);
  cvt(pmem, pmf, 128 * 64);
  cvt(mp2_w, mp2f, 256 * 64);
  cvt(dp1_w, dp1f, 64 * 256);
  cvt(x1_W1, x1w1f, 3 * 16 * 64 * 64);
  cvt(x1_W2, x1w2f, 3 * 16 * 64 * 64);
  cvt(x2_W1, x2w1f, 2 * 16 * 64 * 64);
  cvt(x2_W2, x2w2f, 2 * 16 * 64 * 64);
  cvt(y_W1, yw1f, 2 * 16 * 64 * 64);
  cvt(y_W2, yw2f, 2 * 16 * 64 * 64);
  cvt(xy_W1, xyw1f, 2 * 16 * 128 * 128);
  cvt(xy_W2, xyw2f, 2 * 16 * 128 * 128);
  cvt(x1_V, x1vf, 3 * 16 * 128);
  cvt(x2_V, x2vf, 2 * 16 * 128);
  cvt(y_V, yvf, 2 * 16 * 128);
  cvt(xy_V, xyvf, 2 * 16 * 256);
  cvt(dec_w1, decw1f, DEC * Cc);
  k_pad_dp2<<<(Cc * NDRP + 255) / 256, 256, 0, stream>>>(dp2_w, dp2f);
  k_pack_mwmb<<<(16 * 256 + 255) / 256, 256, 0, stream>>>(mW_w, mb_w, mwmbf);
  k_pack_w2<<<(16 * DEC + 255) / 256, 256, 0, stream>>>(dec_w2, decw2f);

  k_hm_t0<<<(Bb * Cc * Nn + 255) / 256, 256, 0, stream>>>(hidden_x, hm);
  k_embed<<<dim3(BN / 128, 6), 256, (128 * 256 + 128 * 64) * 2, stream>>>(
      embedding, et_w1f, et_b1, et_w2f, et_b2, hm);

  k_memattn<<<BNT / 128, 256, (128 * 256 + 128 * 64 + 128 * 128 + 128 * 64) * 2, stream>>>(
      hm, mp1f, mp1_b, pmf, mp2f, mp2_b, dp1f, dp1_b, res, cts16);

  // NTN calls in reference concat order
  const int t1o[9] = {0, 1, 2, 1, 0, 3, 4, 1, 0};
  const int t2o[9] = {1, 3, 4, 3, 3, 5, 6, 5, 5};
  for (int c = 0; c < 9; ++c) {
    const _Float16 *w1, *w2, *vf;
    const float* bp;
    int dd;
    int fam = (c == 8) ? 0 : (c & 3);           // 0:x1 1:x2 2:y 3:xy
    int idx = (c == 8) ? 2 : (c >> 2);          // which l
    if (fam == 0) {
      w1 = x1w1f + (size_t)idx * 16 * 64 * 64; w2 = x1w2f + (size_t)idx * 16 * 64 * 64;
      vf = x1vf + (size_t)idx * 16 * 128; bp = x1_b + idx * 16; dd = 64;
    } else if (fam == 1) {
      w1 = x2w1f + (size_t)idx * 16 * 64 * 64; w2 = x2w2f + (size_t)idx * 16 * 64 * 64;
      vf = x2vf + (size_t)idx * 16 * 128; bp = x2_b + idx * 16; dd = 64;
    } else if (fam == 2) {
      w1 = yw1f + (size_t)idx * 16 * 64 * 64; w2 = yw2f + (size_t)idx * 16 * 64 * 64;
      vf = yvf + (size_t)idx * 16 * 128; bp = y_b + idx * 16; dd = 64;
    } else {
      w1 = xyw1f + (size_t)idx * 16 * 128 * 128; w2 = xyw2f + (size_t)idx * 16 * 128 * 128;
      vf = xyvf + (size_t)idx * 16 * 256; bp = xy_b + idx * 16; dd = 128;
    }
    if (dd == 64)
      k_ntn<64><<<BN / 128, 256, 2 * 128 * 64 * 2 + 128 * 16 * 4, stream>>>(
          cts16, t1o[c], t2o[c], w1, w2, vf, bp, drifts16, c * 16);
    else
      k_ntn<128><<<BN / 128, 256, 2 * 128 * 128 * 2 + 128 * 16 * 4, stream>>>(
          cts16, t1o[c], t2o[c], w1, w2, vf, bp, drifts16, c * 16);
  }

  k_drift<<<BN / 128, 256, (128 * NDRP + 128 * 256) * 2, stream>>>(
      drifts16, dp2f, dp2_b, mwmbf, mW_b, mb_b, wm15);

  k_dec<<<BN / 128, 256, 128 * 256 * 2 + 128 * 512 * 2 + 128 * 15 * 4, stream>>>(
      hm, res, wm15, decw1f, dec_b1, decw2f, dec_b2, (float*)d_out);
}
